// AtomKpRnnEncoder_80599356277048
// MI455X (gfx1250) — matrix operand order repaired and checked by reading, not compiled, not run on hardware
//
#include <hip/hip_runtime.h>
#include <hip/hip_bf16.h>

typedef __attribute__((ext_vector_type(16))) _Float16 v16h;
typedef __attribute__((ext_vector_type(8)))  float    v8f;
typedef __attribute__((ext_vector_type(4)))  unsigned int v4u;
typedef __attribute__((ext_vector_type(8)))  int      v8i;
typedef __attribute__((ext_vector_type(4)))  int      v4i;

#define B_      32
#define T_      17
#define GRID_   32
#define WMAXK   1344   // >= max Kpad (1312), LDS weight tile row stride bound

#ifndef __has_builtin
#define __has_builtin(x) 0
#endif
#if __has_builtin(__builtin_amdgcn_tensor_load_to_lds)
#define USE_TDM 1
#else
#define USE_TDM 0
#endif

static __device__ __forceinline__ float sigm_(float x) { return 1.f / (1.f + __expf(-x)); }

#if USE_TDM
// Build a 2-D Tensor-DMA descriptor (ISA 8.3/8.4) and DMA [16 x Kpad] f16
// weight rows from global into LDS.  Rows past rows_avail read as zero (TDM
// OOB semantics), which handles Cout < 16 tiles for free.
// This toolchain exposes the 6-operand builtin:
//   (u32x4 g0, i32x8 g1, i32x4 g2, i32x4 g3, i32x8 aux, i32 cpol)
static __device__ __forceinline__ void tdm_load_w(const _Float16* gsrc,
                                                  unsigned lds_off,
                                                  int rows_avail, int Kpad) {
  unsigned long long ga = (unsigned long long)(uintptr_t)gsrc;
  v4u g0;
  g0.x = 1u;                                            // count=1, user mode
  g0.y = lds_off;                                       // lds_addr (bytes)
  g0.z = (unsigned)(ga & 0xffffffffu);                  // global_addr[31:0]
  g0.w = (unsigned)((ga >> 32) & 0x1ffffffu) | (2u << 30);  // addr[56:32], type=2
  unsigned rows = (unsigned)rows_avail;
  v8i g1;
  g1[0] = (int)(1u << 16);                              // wg_mask=0, data_size=1 (2B)
  g1[1] = (int)(((unsigned)Kpad & 0xffffu) << 16);      // tensor_dim0[15:0]
  g1[2] = (int)((((unsigned)Kpad >> 16) & 0xffffu) | ((rows & 0xffffu) << 16));
  g1[3] = (int)(((rows >> 16) & 0xffffu) | (((unsigned)Kpad & 0xffffu) << 16)); // tile_dim0
  g1[4] = 16;                                           // tile_dim1=16, tile_dim2=0
  g1[5] = Kpad;                                         // tensor_dim0_stride[31:0]
  g1[6] = 0;                                            // stride hi / dim1_stride lo
  g1[7] = 0;
  v4i g2 = {0, 0, 0, 0};
  v4i g3 = {0, 0, 0, 0};
  v8i g4 = {0, 0, 0, 0, 0, 0, 0, 0};
  __builtin_amdgcn_tensor_load_to_lds(g0, g1, g2, g3, g4, 0);
}
#endif

// ---------------------------------------------------------------------------
// Implicit-GEMM ConvGRU gates: g = sigmoid(conv3x3x3(cat(x_t, h)) + gb)
// writes rh = r*h (f16) and z (f32).  One wave = one 16(vox) x 16(outch) tile.
// A (voxel patches) 16xK f16 gathered from global; B (weights) Kx16 f16
// staged once into LDS by the Tensor Data Mover.  K = (Cin+Ch)*27 padded.
// ---------------------------------------------------------------------------
__global__ __launch_bounds__(32) void convgru_gates_wmma(
    const _Float16* __restrict__ x, const _Float16* __restrict__ hprev,
    unsigned long long hstride,
    const _Float16* __restrict__ wg, const float* __restrict__ gb,
    _Float16* __restrict__ rh, float* __restrict__ zbuf,
    int t, int Cin, int Ch, int D, int Kpad)
{
  __shared__ _Float16 wlds[16 * WMAXK];
  const int lane = threadIdx.x;
  const int DDD  = D * D * D;
  const int Ktot = (Cin + Ch) * 27;
  const int Cout = 2 * Ch;
  const int m0 = blockIdx.x * 16, n0 = blockIdx.y * 16, b = blockIdx.z;
  const int mv = m0 + (lane & 15);
  const bool mvalid = mv < DDD;
  const int vz = mv / (D * D);
  const int vy = (mv / D) % D;
  const int vx = mv - (mv / D) * D;
  const _Float16* xb = x + ((size_t)(b * T_ + t)) * Cin * DDD;
  const _Float16* hb = hprev + (size_t)b * hstride;
  const int ncol = n0 + (lane & 15);
  const bool nvalid = ncol < Cout;
  const int kbA = (lane & 16) ? 8 : 0;    // A layout: K = e + (e&8) + 8*lanehi
  const int kbB = (lane & 16) ? 8 : 0;    // B layout: K = e + (e&8) + 8*lanehi
  const int wrow = (lane & 15) * Kpad;

#if USE_TDM
  tdm_load_w(wg + (size_t)n0 * Kpad, (unsigned)(uintptr_t)wlds, Cout - n0, Kpad);
  __builtin_amdgcn_s_wait_tensorcnt(0);
#else
  for (int i = lane; i < 16 * Kpad; i += 32) {
    int row = i / Kpad, k = i - row * Kpad;
    wlds[i] = (n0 + row < Cout) ? wg[(size_t)(n0 + row) * Kpad + k] : (_Float16)0.f;
  }
  __syncthreads();
#endif
  __builtin_prefetch(xb, 0, 0);

  v8f acc = {};
  for (int kc = 0; kc < Kpad; kc += 32) {
    v16h a, bm;
#pragma unroll
    for (int e = 0; e < 16; ++e) {
      int K = kc + e + (e & 8) + kbA;
      _Float16 av = (_Float16)0.f;
      if (mvalid && K < Ktot) {
        int ci = K / 27, tap = K - ci * 27;
        int t9 = tap / 9, r9 = tap - t9 * 9;
        int dz = t9 - 1, dy = r9 / 3 - 1, dx = r9 - (r9 / 3) * 3 - 1;
        int zz = vz + dz, yy = vy + dy, xx = vx + dx;
        if ((unsigned)zz < (unsigned)D && (unsigned)yy < (unsigned)D &&
            (unsigned)xx < (unsigned)D) {
          int off = (zz * D + yy) * D + xx;
          av = (ci < Cin) ? xb[(size_t)ci * DDD + off]
                          : hb[(size_t)(ci - Cin) * DDD + off];
        }
      }
      a[e] = av;
      bm[e] = wlds[wrow + kc + e + (e & 8) + kbB];
    }
    acc = __builtin_amdgcn_wmma_f32_16x16x32_f16(false, a, false, bm,
                                                 (short)0, acc, false, false);
  }
  if (!nvalid) return;                      // divergence only after all WMMAs
  const int mbias = (lane & 16) ? 8 : 0;    // D layout: M = r + 8*lanehi
#pragma unroll
  for (int r = 0; r < 8; ++r) {
    int v = m0 + r + mbias;
    if (v < DDD) {
      float g = sigm_(acc[r] + gb[ncol]);
      if (ncol < Ch) {
        float hv = (float)hb[(size_t)ncol * DDD + v];
        rh[((size_t)b * Ch + ncol) * DDD + v] = (_Float16)(g * hv);
      } else {
        zbuf[((size_t)b * Ch + (ncol - Ch)) * DDD + v] = g;
      }
    }
  }
}

// ---------------------------------------------------------------------------
// Candidate conv: n = tanh(conv3x3x3(cat(x_t, r*h)) + cb); h' = (1-z)h + z n
// ---------------------------------------------------------------------------
__global__ __launch_bounds__(32) void convgru_cand_wmma(
    const _Float16* __restrict__ x, const _Float16* __restrict__ rh,
    const _Float16* __restrict__ hprev, unsigned long long hstride,
    const _Float16* __restrict__ wc, const float* __restrict__ cb,
    const float* __restrict__ zbuf, _Float16* __restrict__ hseq,
    int t, int Cin, int Ch, int D, int Kpad)
{
  __shared__ _Float16 wlds[16 * WMAXK];
  const int lane = threadIdx.x;
  const int DDD  = D * D * D;
  const int Ktot = (Cin + Ch) * 27;
  const int m0 = blockIdx.x * 16, n0 = blockIdx.y * 16, b = blockIdx.z;
  const int mv = m0 + (lane & 15);
  const bool mvalid = mv < DDD;
  const int vz = mv / (D * D);
  const int vy = (mv / D) % D;
  const int vx = mv - (mv / D) * D;
  const _Float16* xb  = x + ((size_t)(b * T_ + t)) * Cin * DDD;
  const _Float16* rhb = rh + (size_t)b * Ch * DDD;
  const _Float16* hb  = hprev + (size_t)b * hstride;
  const int ncol = n0 + (lane & 15);
  const bool nvalid = ncol < Ch;
  const int kbA = (lane & 16) ? 8 : 0;
  const int kbB = (lane & 16) ? 8 : 0;
  const int wrow = (lane & 15) * Kpad;

#if USE_TDM
  tdm_load_w(wc + (size_t)n0 * Kpad, (unsigned)(uintptr_t)wlds, Ch - n0, Kpad);
  __builtin_amdgcn_s_wait_tensorcnt(0);
#else
  for (int i = lane; i < 16 * Kpad; i += 32) {
    int row = i / Kpad, k = i - row * Kpad;
    wlds[i] = (n0 + row < Ch) ? wc[(size_t)(n0 + row) * Kpad + k] : (_Float16)0.f;
  }
  __syncthreads();
#endif
  __builtin_prefetch(xb, 0, 0);

  v8f acc = {};
  for (int kc = 0; kc < Kpad; kc += 32) {
    v16h a, bm;
#pragma unroll
    for (int e = 0; e < 16; ++e) {
      int K = kc + e + (e & 8) + kbA;
      _Float16 av = (_Float16)0.f;
      if (mvalid && K < Ktot) {
        int ci = K / 27, tap = K - ci * 27;
        int t9 = tap / 9, r9 = tap - t9 * 9;
        int dz = t9 - 1, dy = r9 / 3 - 1, dx = r9 - (r9 / 3) * 3 - 1;
        int zz = vz + dz, yy = vy + dy, xx = vx + dx;
        if ((unsigned)zz < (unsigned)D && (unsigned)yy < (unsigned)D &&
            (unsigned)xx < (unsigned)D) {
          int off = (zz * D + yy) * D + xx;
          av = (ci < Cin) ? xb[(size_t)ci * DDD + off]
                          : rhb[(size_t)(ci - Cin) * DDD + off];
        }
      }
      a[e] = av;
      bm[e] = wlds[wrow + kc + e + (e & 8) + kbB];
    }
    acc = __builtin_amdgcn_wmma_f32_16x16x32_f16(false, a, false, bm,
                                                 (short)0, acc, false, false);
  }
  if (!nvalid) return;
  const int mbias = (lane & 16) ? 8 : 0;
#pragma unroll
  for (int r = 0; r < 8; ++r) {
    int v = m0 + r + mbias;
    if (v < DDD) {
      float n  = tanhf(acc[r] + cb[ncol]);
      float z  = zbuf[((size_t)b * Ch + ncol) * DDD + v];
      float hp = (float)hb[(size_t)ncol * DDD + v];
      hseq[((size_t)(b * T_ + t) * Ch + ncol) * DDD + v] =
          (_Float16)((1.f - z) * hp + z * n);
    }
  }
}

// --------------------------- small helper kernels ---------------------------
__global__ void prep_w_f16(const float* __restrict__ src, _Float16* __restrict__ dst,
                           int K, int Kpad, int total) {
  int i = blockIdx.x * blockDim.x + threadIdx.x;
  if (i >= total) return;
  int co = i / Kpad, k = i - co * Kpad;
  dst[i] = (k < K) ? (_Float16)src[(size_t)co * K + k] : (_Float16)0.f;
}

__global__ void prep_x1(const float* __restrict__ kps, _Float16* __restrict__ x1,
                        long total) {
  long i = (long)blockIdx.x * blockDim.x + threadIdx.x;
  if (i >= total) return;
  int v = (int)(i % 32768);
  int t = (int)((i / 32768) % T_);
  int b = (int)(i / (32768L * T_));
  float val = (t == 0) ? 1.f : kps[((size_t)b * 16 + (t - 1)) * 32768 + v];
  x1[i] = (_Float16)val;
}

__global__ void make_adj(const int* __restrict__ bonds, float* __restrict__ adjn) {
  int b = blockIdx.x, i = threadIdx.y, j = threadIdx.x;
  __shared__ float a[T_][T_];
  __shared__ float rs[T_];
  float v = ((bonds[((size_t)b * T_ + i) * T_ + j] > 0) ? 1.f : 0.f) +
            ((i == j) ? 1.f : 0.f);
  a[i][j] = v;
  __syncthreads();
  if (j == 0) { float s = 0.f; for (int k = 0; k < T_; ++k) s += a[i][k]; rs[i] = s; }
  __syncthreads();
  adjn[((size_t)b * T_ + i) * T_ + j] = a[i][j] / rs[i];
}

// input-projection for GRU1 (embedding lookup fused)
__global__ void gru_xw_tok(const int* __restrict__ atom_types,
                           const float* __restrict__ emb,
                           const float* __restrict__ w_ih,
                           const float* __restrict__ b_ih,
                           float* __restrict__ xw) {
  int m = blockIdx.x, g = threadIdx.x;          // m in [0,544), g in [0,384)
  int b = m / T_, t = m - b * T_;
  int tok = (t == 0) ? 1 : atom_types[b * 16 + (t - 1)];   // ATOM_START = 1
  const float* e = emb + (size_t)tok * 64;
  const float* w = w_ih + (size_t)g * 64;
  float s = b_ih[g];
  for (int k = 0; k < 64; ++k) s += e[k] * w[k];
  xw[(size_t)m * 384 + g] = s;
}

// input-projection for GRU2 (dense 128-wide input)
__global__ void gru_xw_seq(const float* __restrict__ in,
                           const float* __restrict__ w_ih,
                           const float* __restrict__ b_ih,
                           float* __restrict__ xw) {
  int m = blockIdx.x, g = threadIdx.x;
  const float* e = in + (size_t)m * 128;
  const float* w = w_ih + (size_t)g * 128;
  float s = b_ih[g];
  for (int k = 0; k < 128; ++k) s += e[k] * w[k];
  xw[(size_t)m * 384 + g] = s;
}

// torch-order GRU scan (r,z,n), H=128, T=17; one block per batch element
__global__ void gru_scan(const float* __restrict__ xw, const float* __restrict__ w_hh,
                         const float* __restrict__ b_hh, float* __restrict__ outseq,
                         float* __restrict__ hout) {
  const int H = 128;
  int b = blockIdx.x, tid = threadIdx.x;
  __shared__ float h[H];
  h[tid] = 0.f;
  __syncthreads();
  for (int t = 0; t < T_; ++t) {
    const float* xt = xw + ((size_t)b * T_ + t) * 3 * H;
    float hr = b_hh[tid], hz = b_hh[H + tid], hn = b_hh[2 * H + tid];
    const float* wr = w_hh + (size_t)tid * H;
    const float* wz = w_hh + (size_t)(H + tid) * H;
    const float* wn = w_hh + (size_t)(2 * H + tid) * H;
    for (int k = 0; k < H; ++k) {
      float hk = h[k];
      hr += wr[k] * hk; hz += wz[k] * hk; hn += wn[k] * hk;
    }
    float r = sigm_(xt[tid] + hr);
    float z = sigm_(xt[H + tid] + hz);
    float n = tanhf(xt[2 * H + tid] + r * hn);
    float hnew = (1.f - z) * n + z * h[tid];
    __syncthreads();
    h[tid] = hnew;
    __syncthreads();
    if (outseq) outseq[((size_t)b * T_ + t) * H + tid] = hnew;
  }
  if (hout) hout[(size_t)b * H + tid] = h[tid];
}

// bond attention on a [B,17,128] f32 sequence
__global__ void attn_f32(const float* __restrict__ in, const float* __restrict__ adjn,
                         float* __restrict__ out) {
  int b = blockIdx.x, i = blockIdx.y, f = threadIdx.x;
  float s = 0.f;
  for (int j = 0; j < T_; ++j)
    s += adjn[((size_t)b * T_ + i) * T_ + j] * in[((size_t)b * T_ + j) * 128 + f];
  out[((size_t)b * T_ + i) * 128 + f] = s;
}

// deterministic BN stats, stage 1: per-(b,t,c) slab partial sum / sumsq
__global__ void bn_stats(const _Float16* __restrict__ h, float* __restrict__ part,
                         int DDD) {
  int slab = blockIdx.x, tid = threadIdx.x;
  const _Float16* p = h + (size_t)slab * DDD;
  float s = 0.f, s2 = 0.f;
  for (int v = tid; v < DDD; v += blockDim.x) {
    float x = (float)p[v];
    s += x; s2 += x * x;
  }
  __shared__ float sh[256], sh2[256];
  sh[tid] = s; sh2[tid] = s2;
  __syncthreads();
  for (int st = 128; st > 0; st >>= 1) {
    if (tid < st) { sh[tid] += sh[tid + st]; sh2[tid] += sh2[tid + st]; }
    __syncthreads();
  }
  if (tid == 0) { part[(size_t)slab * 2] = sh[0]; part[(size_t)slab * 2 + 1] = sh2[0]; }
}

// stage 2: per-channel mean/var -> affine (scale, shift)
__global__ void bn_finalize(const float* __restrict__ part,
                            const float* __restrict__ gamma,
                            const float* __restrict__ beta,
                            float* __restrict__ scale, float* __restrict__ shift,
                            int Ch, float cnt) {
  int c = threadIdx.x;
  float s = 0.f, s2 = 0.f;
  for (int bt = 0; bt < B_ * T_; ++bt) {
    s  += part[((size_t)bt * Ch + c) * 2];
    s2 += part[((size_t)bt * Ch + c) * 2 + 1];
  }
  float m = s / cnt;
  float var = s2 / cnt - m * m;
  float sc = gamma[c] * rsqrtf(var + 1e-5f);
  scale[c] = sc;
  shift[c] = beta[c] - m * sc;
}

// BN-normalize + 2x2x2 max-pool
__global__ void bn_pool(const _Float16* __restrict__ h, const float* __restrict__ scale,
                        const float* __restrict__ shift, _Float16* __restrict__ pooled,
                        int Ch, int D, int D2, long total) {
  long i = (long)blockIdx.x * blockDim.x + threadIdx.x;
  if (i >= total) return;
  int VP = D2 * D2 * D2;
  int vp = (int)(i % VP);
  int c  = (int)((i / VP) % Ch);
  long s = i / ((long)VP * Ch);                 // bt index
  int z2 = vp / (D2 * D2), y2 = (vp / D2) % D2, x2 = vp % D2;
  const _Float16* base = h + ((size_t)s * Ch + c) * D * D * D;
  float sc = scale[c], sf = shift[c];
  float mx = -3.4e38f;
  for (int dz = 0; dz < 2; ++dz)
    for (int dy = 0; dy < 2; ++dy)
      for (int dx = 0; dx < 2; ++dx) {
        int v = ((2 * z2 + dz) * D + 2 * y2 + dy) * D + 2 * x2 + dx;
        float val = sc * (float)base[v] + sf;
        mx = fmaxf(mx, val);
      }
  pooled[i] = (_Float16)mx;
}

// bond attention over the atom/time axis of pooled [B,17,Ch,VP] -> next x (f16)
__global__ void attn_mix(const _Float16* __restrict__ pooled,
                         const float* __restrict__ adjn, _Float16* __restrict__ xnext,
                         int Ch, int VP, long total) {
  long idx = (long)blockIdx.x * blockDim.x + threadIdx.x;
  if (idx >= total) return;
  int vp = (int)(idx % VP);
  int c  = (int)((idx / VP) % Ch);
  int i  = (int)((idx / ((long)VP * Ch)) % T_);
  int b  = (int)(idx / ((long)VP * Ch * T_));
  float s = 0.f;
  for (int j = 0; j < T_; ++j)
    s += adjn[((size_t)b * T_ + i) * T_ + j] *
         (float)pooled[(((size_t)b * T_ + j) * Ch + c) * VP + vp];
  xnext[idx] = (_Float16)s;
}

// final FC [B,160] -> [B,1024] with leaky-relu
__global__ void fc_leaky(const _Float16* __restrict__ xlast, const float* __restrict__ hidden,
                         const float* __restrict__ fw, const float* __restrict__ fb,
                         float* __restrict__ out) {
  int b = blockIdx.x, tid = threadIdx.x;
  __shared__ float in[160];
  if (tid < 32)       in[tid] = (float)xlast[((size_t)b * T_ + 16) * 32 + tid];
  else if (tid < 160) in[tid] = hidden[(size_t)b * 128 + (tid - 32)];
  __syncthreads();
  for (int o = tid; o < 1024; o += 256) {
    const float* w = fw + (size_t)o * 160;
    float s = fb[o];
    for (int k = 0; k < 160; ++k) s += w[k] * in[k];
    out[(size_t)b * 1024 + o] = (s >= 0.f) ? s : 0.1f * s;
  }
}

// ---------------------------------------------------------------------------
extern "C" void kernel_launch(void* const* d_in, const int* in_sizes, int n_in,
                              void* d_out, int out_size, void* d_ws, size_t ws_size,
                              hipStream_t stream) {
  (void)n_in; (void)out_size; (void)ws_size;
  const int FIL[5] = {2, 4, 8, 16, 32};

  const int*   atom_types = (const int*)d_in[0];
  const int*   bonds      = (const int*)d_in[1];
  const float* kps        = (const float*)d_in[2];
  const float* embedding  = (const float*)d_in[3];
  const float* atn_w_ih   = (const float*)d_in[4];
  const float* atn_w_hh   = (const float*)d_in[5];
  const float* atn_b_ih   = (const float*)d_in[6];
  const float* atn_b_hh   = (const float*)d_in[7];
  const float* atom_w_ih  = (const float*)d_in[8];
  const float* atom_w_hh  = (const float*)d_in[9];
  const float* atom_b_ih  = (const float*)d_in[10];
  const float* atom_b_hh  = (const float*)d_in[11];

  // kp_params / bn_params flattening order: detect insertion vs key-sorted
  const float *gw[5], *gbp[5], *cwp[5], *cbp[5], *bng[5], *bnb[5];
  int idx = 12;
  bool ins = (in_sizes[12] == 2 * FIL[0] * (1 + FIL[0]) * 27);  // gw L1 = 324
  for (int L = 0; L < 5; ++L) {
    if (ins) {
      gw[L]  = (const float*)d_in[idx++]; gbp[L] = (const float*)d_in[idx++];
      cwp[L] = (const float*)d_in[idx++]; cbp[L] = (const float*)d_in[idx++];
    } else {  // alphabetical: cb, cw, gb, gw
      cbp[L] = (const float*)d_in[idx++]; cwp[L] = (const float*)d_in[idx++];
      gbp[L] = (const float*)d_in[idx++]; gw[L]  = (const float*)d_in[idx++];
    }
  }
  for (int L = 0; L < 5; ++L) {
    if (ins) { bng[L] = (const float*)d_in[idx++]; bnb[L] = (const float*)d_in[idx++]; }
    else     { bnb[L] = (const float*)d_in[idx++]; bng[L] = (const float*)d_in[idx++]; }
  }
  const float* fc_w = (const float*)d_in[idx++];
  const float* fc_b = (const float*)d_in[idx++];
  float* out = (float*)d_out;

  // ----- workspace layout -----
  char* wsb = (char*)d_ws;
  size_t off = 0;
  auto alloc = [&](size_t bytes) -> char* {
    off = (off + 255) & ~(size_t)255;
    char* p = wsb + off;
    off += bytes;
    return p;
  };
  _Float16* xA     = (_Float16*)alloc((size_t)544 * 32768 * 2);     // level inputs (odd)
  _Float16* xB     = (_Float16*)alloc((size_t)544 * 2 * 4096 * 2);  // level inputs (even)
  _Float16* hseq   = (_Float16*)alloc((size_t)544 * 2 * 32768 * 2); // per-level hidden seq
  _Float16* rhb    = (_Float16*)alloc((size_t)32 * 2 * 32768 * 2);
  float*    zb     = (float*)   alloc((size_t)32 * 2 * 32768 * 4);
  _Float16* zero16 = (_Float16*)alloc((size_t)65536 * 2);
  _Float16* pooled = (_Float16*)alloc((size_t)544 * 2 * 4096 * 2);
  float*    part   = (float*)   alloc((size_t)B_ * T_ * 32 * 2 * 4);
  float*    scale  = (float*)   alloc(64 * 4);
  float*    shift  = (float*)   alloc(64 * 4);
  float*    adjn   = (float*)   alloc((size_t)B_ * T_ * T_ * 4);
  float*    xw     = (float*)   alloc((size_t)544 * 384 * 4);
  float*    seqA   = (float*)   alloc((size_t)544 * 128 * 4);
  float*    seqB   = (float*)   alloc((size_t)544 * 128 * 4);
  float*    hidden = (float*)   alloc((size_t)32 * 128 * 4);
  _Float16* wg16[5]; _Float16* wc16[5];
  int KpadL[5];
  {
    int fp = 1;
    for (int L = 0; L < 5; ++L) {
      int f = FIL[L], C = fp + f, K = C * 27, Kp = (K + 31) & ~31;
      KpadL[L] = Kp;
      wg16[L] = (_Float16*)alloc((size_t)2 * f * Kp * 2);
      wc16[L] = (_Float16*)alloc((size_t)f * Kp * 2);
      fp = f;
    }
  }

  (void)hipMemsetAsync(zero16, 0, (size_t)65536 * 2, stream);

  // weight conversion f32 -> padded f16 rows
  {
    int fp = 1;
    for (int L = 0; L < 5; ++L) {
      int f = FIL[L], C = fp + f, K = C * 27, Kp = KpadL[L];
      int tot = 2 * f * Kp;
      prep_w_f16<<<(tot + 255) / 256, 256, 0, stream>>>(gw[L], wg16[L], K, Kp, tot);
      tot = f * Kp;
      prep_w_f16<<<(tot + 255) / 256, 256, 0, stream>>>(cwp[L], wc16[L], K, Kp, tot);
      fp = f;
    }
  }
  // level-1 input (ones plane at t=0, kps after)
  {
    long tot = 544L * 32768;
    prep_x1<<<(unsigned)((tot + 255) / 256), 256, 0, stream>>>(kps, xA, tot);
  }
  // normalized adjacency
  make_adj<<<B_, dim3(T_, T_), 0, stream>>>(bonds, adjn);

  // ----- atom path (tiny, latency-bound: scalar) -----
  gru_xw_tok<<<544, 384, 0, stream>>>(atom_types, embedding, atn_w_ih, atn_b_ih, xw);
  gru_scan<<<B_, 128, 0, stream>>>(xw, atn_w_hh, atn_b_hh, seqA, (float*)nullptr);
  attn_f32<<<dim3(B_, T_), 128, 0, stream>>>(seqA, adjn, seqB);
  gru_xw_seq<<<544, 384, 0, stream>>>(seqB, atom_w_ih, atom_b_ih, xw);
  gru_scan<<<B_, 128, 0, stream>>>(xw, atom_w_hh, atom_b_hh, (float*)nullptr, hidden);

  // ----- ConvGRU pyramid (WMMA implicit GEMM, TDM-staged weights) -----
  {
    int fp = 1;
    for (int L = 0; L < 5; ++L) {
      int f = FIL[L], Cin = fp, Ch = f, D = GRID_ >> L, DDD = D * D * D;
      int Kp = KpadL[L];
      const _Float16* xc = (L & 1) ? xB : xA;
      _Float16* xn = (L & 1) ? xA : xB;
      for (int t = 0; t < T_; ++t) {
        const _Float16* hp = (t == 0) ? zero16 : hseq + (size_t)(t - 1) * Ch * DDD;
        unsigned long long hs = (t == 0) ? 0ULL : (unsigned long long)T_ * Ch * DDD;
        dim3 gg((DDD + 15) / 16, (2 * Ch + 15) / 16, B_);
        convgru_gates_wmma<<<gg, 32, 0, stream>>>(xc, hp, hs, wg16[L], gbp[L],
                                                  rhb, zb, t, Cin, Ch, D, Kp);
        dim3 gc((DDD + 15) / 16, (Ch + 15) / 16, B_);
        convgru_cand_wmma<<<gc, 32, 0, stream>>>(xc, rhb, hp, hs, wc16[L], cbp[L],
                                                 zb, hseq, t, Cin, Ch, D, Kp);
      }
      bn_stats<<<B_ * T_ * Ch, 256, 0, stream>>>(hseq, part, DDD);
      bn_finalize<<<1, Ch, 0, stream>>>(part, bng[L], bnb[L], scale, shift, Ch,
                                        (float)((double)B_ * T_ * DDD));
      int D2 = D / 2, VP = D2 * D2 * D2;
      long tot = 544L * Ch * VP;
      bn_pool<<<(unsigned)((tot + 255) / 256), 256, 0, stream>>>(hseq, scale, shift,
                                                                 pooled, Ch, D, D2, tot);
      attn_mix<<<(unsigned)((tot + 255) / 256), 256, 0, stream>>>(pooled, adjn, xn,
                                                                  Ch, VP, tot);
      fp = f;
    }
  }

  // final level output lives in xB: [B,17,32,1]; concat with hidden -> FC
  fc_leaky<<<B_, 256, 0, stream>>>(xB, hidden, fc_w, fc_b, out);
}
